// RelationGAT_36077725286633
// MI455X (gfx1250) — compile-verified, matrix operand order checked by reading, not run on hardware
//
#include <hip/hip_runtime.h>

// ---------------------------------------------------------------------------
// RelationGAT on MI455X (gfx1250):
//   out = rownorm( D @ (A @ sum_s item_embedding[items]) )
// Re-associated to avoid the 4096^3 GEMM (137 GFLOP -> 8.6 GFLOP).
// GEMMs use V_WMMA_F32_16X16X32_BF16 (bf16 in, f32 accumulate).
// ---------------------------------------------------------------------------

typedef __attribute__((ext_vector_type(16))) __bf16 v16bf;
typedef __attribute__((ext_vector_type(8)))  float  v8f;

#define HIDDEN 128
#define BATCH  4096
#define SESS   50
#define KC     256    // K-chunk staged in LDS per workgroup iteration
#define LDA    264    // padded LDS row stride (bf16 elems): 528B -> 4-bank shift/row

__device__ __forceinline__ unsigned short f2bf(float f) {
    unsigned int u = __float_as_uint(f);
    u += 0x7FFFu + ((u >> 16) & 1u);          // round-to-nearest-even
    return (unsigned short)(u >> 16);
}

// ---------------------------------------------------------------------------
// Kernel 1: gather + session-sum, write E^T (HIDDEN x BATCH) in bf16 so the
// GEMM B-operand is contiguous along K. Block = 128 threads (one per h),
// each block handles 16 batch rows -> coalesced 512B embedding reads across
// the 128 lanes, and each thread writes one contiguous 32B run of E^T.
// ---------------------------------------------------------------------------
__global__ __launch_bounds__(128) void gather_sum_kernel(
    const float* __restrict__ emb,          // (100000, 128) f32
    const int*   __restrict__ items,        // (4096, 50) int
    unsigned short* __restrict__ Et)        // (128, 4096) bf16
{
    const int h  = threadIdx.x;             // 0..127
    const int b0 = blockIdx.x * 16;

    __align__(16) unsigned short vals[16];
    for (int i = 0; i < 16; ++i) {
        const int* it = items + (long)(b0 + i) * SESS;
        float acc = 0.f;
        #pragma unroll 5
        for (int s = 0; s < SESS; ++s) {
            acc += emb[(long)it[s] * HIDDEN + h];
        }
        vals[i] = f2bf(acc);
    }
    unsigned short* dst = Et + (long)h * BATCH + b0;   // 32B aligned
    *(uint4*)(dst)     = *(const uint4*)(vals);
    *(uint4*)(dst + 8) = *(const uint4*)(vals + 8);
}

// ---------------------------------------------------------------------------
// GEMM: Out(4096x128) = L(4096x4096 f32) @ B, with B given transposed as
// Bt (128x4096 bf16). Workgroup = 256 threads = 8 waves; owns a 16x128 output
// tile (16 rows of L shared by all waves via LDS; wave w owns N-tile w*16).
// K-loop: V_WMMA_F32_16X16X32_BF16, fragments per ISA 7.12.2 striping.
//   MODE 0: write Out^T (128x4096) bf16  (feeds the next GEMM's B operand)
//   MODE 1: fuse row-wise L2 normalization, write f32 (4096x128)
// ---------------------------------------------------------------------------
template <int MODE>
__global__ __launch_bounds__(256) void mm_kernel(
    const float* __restrict__ L,             // 4096x4096 f32 row-major
    const unsigned short* __restrict__ Bt,   // 128x4096 bf16 (output-col major)
    unsigned short* __restrict__ Ot,         // MODE 0: 128x4096 bf16
    float* __restrict__ out)                 // MODE 1: 4096x128 f32
{
    __shared__ unsigned short sA[16 * LDA];  // staged A chunk (bf16)
    __shared__ float sOut[16 * HIDDEN];      // MODE 1: full output tile
    __shared__ float sNorm[16];

    const int tid   = threadIdx.x;
    const int wave  = tid >> 5;
    const int lane  = tid & 31;
    const int m0    = blockIdx.x * 16;       // output row tile
    const int n0    = wave * 16;             // this wave's output col tile
    const int lrow  = lane & 15;
    const int lhalf = (lane >> 4) * 8;       // K-offset of upper half-wave

    // cooperative A-staging coords: thread -> (row, 16-col run)
    const int srow = tid >> 4;               // 0..15
    const int scol = (tid & 15) * 16;        // 0..240
    const float* lptr = L + (long)(m0 + srow) * BATCH + scol;

    const unsigned short* bptr = Bt + (long)(n0 + lrow) * BATCH;

    v8f acc = {};                            // f32 accumulator (C/D 16x16)

    for (int kc = 0; kc < BATCH; kc += KC) {
        // ---- stage 16xKC chunk of L into LDS as bf16 (coalesced 64B/thread)
        {
            const float* src = lptr + kc;
            if (kc + KC < BATCH) __builtin_prefetch(src + KC, 0, 1);
            float4 f0 = *(const float4*)(src);
            float4 f1 = *(const float4*)(src + 4);
            float4 f2 = *(const float4*)(src + 8);
            float4 f3 = *(const float4*)(src + 12);
            __align__(16) unsigned short t[16];
            t[0]=f2bf(f0.x); t[1]=f2bf(f0.y); t[2]=f2bf(f0.z); t[3]=f2bf(f0.w);
            t[4]=f2bf(f1.x); t[5]=f2bf(f1.y); t[6]=f2bf(f1.z); t[7]=f2bf(f1.w);
            t[8]=f2bf(f2.x); t[9]=f2bf(f2.y); t[10]=f2bf(f2.z); t[11]=f2bf(f2.w);
            t[12]=f2bf(f3.x); t[13]=f2bf(f3.y); t[14]=f2bf(f3.z); t[15]=f2bf(f3.w);
            unsigned short* d = &sA[srow * LDA + scol];   // 16B aligned
            *(uint4*)(d)     = *(const uint4*)(t);
            *(uint4*)(d + 8) = *(const uint4*)(t + 8);
        }
        __syncthreads();

        // ---- 8 WMMAs over this chunk
        #pragma unroll
        for (int j = 0; j < KC / 32; ++j) {
            const int k0 = j * 32;
            union { v16bf v; uint4 q[2]; } fa, fb;
            // A fragment: lane holds row lrow, K = k0+lhalf+0..7 and +16..23
            const unsigned short* ap = &sA[lrow * LDA + k0 + lhalf];
            fa.q[0] = *(const uint4*)(ap);
            fa.q[1] = *(const uint4*)(ap + 16);
            // B fragment: lane holds col n0+lrow, same K striping, from B^T
            const unsigned short* bp = bptr + kc + k0 + lhalf;
            fb.q[0] = *(const uint4*)(bp);
            fb.q[1] = *(const uint4*)(bp + 16);
            acc = __builtin_amdgcn_wmma_f32_16x16x32_bf16(
                false, fa.v, false, fb.v, (short)0, acc, false, false);
        }
        __syncthreads();
    }

    if (MODE == 0) {
        // D-frag: lane holds col n = n0+lrow, rows m0 + 8*(lane>>4) + r.
        // Those 8 rows are contiguous in Out^T -> one 16B store per lane.
        __align__(16) unsigned short pk[8];
        #pragma unroll
        for (int r = 0; r < 8; ++r) pk[r] = f2bf(acc[r]);
        unsigned short* dst =
            Ot + (long)(n0 + lrow) * BATCH + m0 + (lane >> 4) * 8;
        *(uint4*)dst = *(const uint4*)pk;
    } else {
        // fused row-wise L2 normalization over the full 16x128 tile
        #pragma unroll
        for (int r = 0; r < 8; ++r) {
            const int m = (lane >> 4) * 8 + r;
            sOut[m * HIDDEN + n0 + lrow] = acc[r];
        }
        __syncthreads();
        if (tid < 16) {
            float s = 0.f;
            #pragma unroll 8
            for (int n = 0; n < HIDDEN; ++n) {
                const float v = sOut[tid * HIDDEN + n];
                s += v * v;
            }
            sNorm[tid] = rsqrtf(s);
        }
        __syncthreads();
        #pragma unroll
        for (int e = tid; e < 16 * HIDDEN; e += 256) {
            const int m = e >> 7, n = e & (HIDDEN - 1);
            out[(long)(m0 + m) * HIDDEN + n] = sOut[e] * sNorm[m];
        }
    }
}

// ---------------------------------------------------------------------------
// kernel_launch
//   d_in[0] item_embedding (100000,128) f32
//   d_in[1] items          (4096,50)    int
//   d_in[2] A              (4096,4096)  f32
//   d_in[3] D              (4096,4096)  f32
//   d_in[4] target_embedding             (unused by reference)
//   d_out   (4096,128) f32
//   d_ws    scratch: E^T (1MB bf16) + T^T (1MB bf16)
// ---------------------------------------------------------------------------
extern "C" void kernel_launch(void* const* d_in, const int* in_sizes, int n_in,
                              void* d_out, int out_size, void* d_ws, size_t ws_size,
                              hipStream_t stream) {
    const float* emb   = (const float*)d_in[0];
    const int*   items = (const int*)d_in[1];
    const float* A     = (const float*)d_in[2];
    const float* D     = (const float*)d_in[3];
    float*       out   = (float*)d_out;

    unsigned short* Et = (unsigned short*)d_ws;                  // 128x4096 bf16
    unsigned short* Tt = Et + (size_t)HIDDEN * BATCH;            // 128x4096 bf16

    // 1) E^T = (sum_s item_embedding[items])^T   (bf16)
    gather_sum_kernel<<<BATCH / 16, 128, 0, stream>>>(emb, items, Et);
    // 2) T^T = (A @ E)^T                          (bf16, WMMA)
    mm_kernel<0><<<BATCH / 16, 256, 0, stream>>>(A, Et, Tt, nullptr);
    // 3) out = rownorm(D @ T)                     (f32 out, WMMA + fused norm)
    mm_kernel<1><<<BATCH / 16, 256, 0, stream>>>(D, Tt, nullptr, out);
}
